// ourLIFSALayer_60876866453587
// MI455X (gfx1250) — matrix operand order repaired and checked by reading, not compiled, not run on hardware
//
#include <hip/hip_runtime.h>

// ---------------------------------------------------------------------------
// Problem constants (from reference setup_inputs)
// ---------------------------------------------------------------------------
#define BB 16
#define TT 512
#define HH 512
#define NHD 8
#define DH 64           // head dim
#define MT (BB*TT)      // 8192 rows

typedef __attribute__((ext_vector_type(16))) __bf16 v16bf;
typedef __attribute__((ext_vector_type(8)))  __bf16 v8bf;
typedef __attribute__((ext_vector_type(8)))  float  v8f;
typedef __attribute__((ext_vector_type(4)))  float  v4f;
typedef __attribute__((ext_vector_type(4)))  unsigned short v4us;
typedef __attribute__((ext_vector_type(8)))  unsigned short v8us;

// float -> bf16 bits, round-to-nearest-even
static __device__ inline unsigned short f2b(float f) {
  unsigned u = __builtin_bit_cast(unsigned, f);
  unsigned r = u + 0x7FFFu + ((u >> 16) & 1u);
  return (unsigned short)(r >> 16);
}

static __device__ inline v8bf ld8(const unsigned short* p) {
  return *(const v8bf*)p;   // 16B aligned by construction
}

static __device__ inline v16bf mk16(v8bf lo, v8bf hi) {
  v16bf r;
#pragma unroll
  for (int i = 0; i < 8; ++i) { r[i] = lo[i]; r[8 + i] = hi[i]; }
  return r;
}

static __device__ inline v8f wmma_bf16(v16bf a, v16bf b, v8f c) {
  // emits v_wmma_f32_16x16x32_bf16
  return __builtin_amdgcn_wmma_f32_16x16x32_bf16(
      /*neg_a=*/false, a, /*neg_b=*/false, b,
      /*c_mod=*/(short)0, c, /*reuse_a=*/false, /*reuse_b=*/false);
}

// ---------------------------------------------------------------------------
// Generic GEMM:  C[M,N] = A[M,K] * B[K,N]  (+ optional residual)
// All configuration is compile-time: no runtime branches in hot loops.
// Block = 256 threads (8 waves). Tile 128x128x32. Wave computes 64x32
// via 4(M) x 2(N) accumulator fragments -> 8 WMMA per K-step.
// ---------------------------------------------------------------------------
template <int ABF, int HASRES, int OUTF32, int OUTBF>
__global__ __launch_bounds__(256)
void gemm_bf16_wmma(const void* __restrict__ Aptr,
                    const float* __restrict__ Bw,
                    const float* __restrict__ Res,
                    float* __restrict__ Cf,
                    unsigned short* __restrict__ Cbf,
                    int N, int K) {
  __shared__ unsigned short Asm[128 * 32];   // [m][k]
  __shared__ unsigned short Bsm[128 * 32];   // transposed: [n][k]

  const int tid = threadIdx.x;
  const int wid = tid >> 5, lane = tid & 31;
  const int lh = lane & 15, lq = lane >> 4;      // lane%16, lane/16
  const int bm0 = blockIdx.y * 128, bn0 = blockIdx.x * 128;
  const int wm0 = (wid >> 2) * 64;               // 0 or 64
  const int wn0 = (wid & 3) * 32;                // 0,32,64,96

  const float* Af = (const float*)Aptr;
  const unsigned short* Ab = (const unsigned short*)Aptr;

  v8f acc[4][2];
#pragma unroll
  for (int mf = 0; mf < 4; ++mf)
#pragma unroll
    for (int nf = 0; nf < 2; ++nf) acc[mf][nf] = (v8f)(0.0f);

  // staging coordinates (vectorized: 4 elements per op)
  const int ar = (tid << 2) >> 5;                // A: row 0..31 step over i
  const int ac4 = (tid & 7) * 4;                 // A: col {0,4,...,28}
  const int bk = (tid << 2) >> 7;                // B: k 0..7 step over i
  const int bn4 = (tid & 31) * 4;                // B: n {0,...,124}

  for (int k0 = 0; k0 < K; k0 += 32) {
    __syncthreads();
    // --- stage A tile: float4 / ushort4 loads, b64 LDS stores ---
#pragma unroll
    for (int i = 0; i < 4; ++i) {
      int r = ar + i * 32;                       // 0..127
      v4us o;
      if (ABF) {
        o = *(const v4us*)(Ab + (size_t)(bm0 + r) * K + k0 + ac4);
      } else {
        v4f vv = *(const v4f*)(Af + (size_t)(bm0 + r) * K + k0 + ac4);
#pragma unroll
        for (int j = 0; j < 4; ++j) o[j] = f2b(vv[j]);
      }
      *(v4us*)(&Asm[r * 32 + ac4]) = o;
    }
    // --- stage B tile transposed: coalesced float4 loads ---
#pragma unroll
    for (int i = 0; i < 4; ++i) {
      int k = bk + i * 8;                        // 0..31
      v4f vv = *(const v4f*)(Bw + (size_t)(k0 + k) * N + bn0 + bn4);
#pragma unroll
      for (int j = 0; j < 4; ++j) Bsm[(bn4 + j) * 32 + k] = f2b(vv[j]);
    }
    // prefetch next K-tile (emits global_prefetch_b8)
    if (k0 + 32 < K) {
      if (ABF)
        __builtin_prefetch(Ab + (size_t)(bm0 + ar) * K + k0 + 32 + ac4, 0, 0);
      else
        __builtin_prefetch(Af + (size_t)(bm0 + ar) * K + k0 + 32 + ac4, 0, 0);
      __builtin_prefetch(Bw + (size_t)(k0 + 32 + bk) * N + bn0 + bn4, 0, 0);
    }
    __syncthreads();

    // B fragments (32x16 bf16 layout: N=lane%16, K = 8*(lane/16)+e / +16)
    v16bf bfrag[2];
#pragma unroll
    for (int nf = 0; nf < 2; ++nf) {
      const unsigned short* bp = &Bsm[(wn0 + nf * 16 + lh) * 32 + lq * 8];
      bfrag[nf] = mk16(ld8(bp), ld8(bp + 16));
    }
    // A fragments (16x32 bf16 layout: M=lane%16, K = 8*(lane/16)+e / +16)
#pragma unroll
    for (int mf = 0; mf < 4; ++mf) {
      const unsigned short* ap = &Asm[(wm0 + mf * 16 + lh) * 32 + lq * 8];
      v16bf afrag = mk16(ld8(ap), ld8(ap + 16));
#pragma unroll
      for (int nf = 0; nf < 2; ++nf)
        acc[mf][nf] = wmma_bf16(afrag, bfrag[nf], acc[mf][nf]);
    }
  }

  // epilogue: C layout row = r + 8*(lane/16), col = lane%16
#pragma unroll
  for (int mf = 0; mf < 4; ++mf)
#pragma unroll
    for (int nf = 0; nf < 2; ++nf)
#pragma unroll
      for (int r = 0; r < 8; ++r) {
        int row = bm0 + wm0 + mf * 16 + r + lq * 8;
        int col = bn0 + wn0 + nf * 16 + lh;
        size_t off = (size_t)row * N + col;
        float v = acc[mf][nf][r];
        if (HASRES) v += Res[off];
        if (OUTF32) Cf[off] = v;
        if (OUTBF)  Cbf[off] = f2b(v);
      }
}

// ---------------------------------------------------------------------------
// Flash attention over bf16 q/k/v, per (b, h, 128 query rows).
// 8 waves x 16 query rows. Key tiles of 32. 9 WMMA per tile
// (4 S, 4 O, 1 row-sum-of-P via ones matrix: sum shuffles -> matrix pipe).
// ---------------------------------------------------------------------------
__global__ __launch_bounds__(256)
void flash_attn_wmma(const unsigned short* __restrict__ Q,
                     const unsigned short* __restrict__ Kb,
                     const unsigned short* __restrict__ Vb,
                     unsigned short* __restrict__ O) {
  __shared__ unsigned short vT[DH * 32];        // [dv][key]  4 KB
  __shared__ unsigned short pbuf[8 * 16 * 32];  // per-wave P  8 KB

  const int tid = threadIdx.x;
  const int wid = tid >> 5, lane = tid & 31;
  const int lh = lane & 15, lq = lane >> 4;
  const int b = blockIdx.z, h = blockIdx.y;
  const int q0 = blockIdx.x * 128 + wid * 16;
  const size_t base = ((size_t)b * TT) * HH + (size_t)h * DH;

  // q A-fragments: rows q0+lane%16, K = head dim (two 16x32 slabs)
  const unsigned short* qp = Q + base + (size_t)(q0 + lh) * HH + lq * 8;
  const v16bf aq0 = mk16(ld8(qp), ld8(qp + 16));        // d 0..31
  const v16bf aq1 = mk16(ld8(qp + 32), ld8(qp + 48));   // d 32..63

  // all-ones B fragment: P x 1 -> row sums broadcast in C layout
  v16bf onesB;
#pragma unroll
  for (int i = 0; i < 16; ++i) onesB[i] = (__bf16)1.0f;

  // V staging coordinates: each thread moves 8 contiguous dv of one key row
  const int vkey = tid >> 3;                    // 0..31
  const int vdv8 = (tid & 7) * 8;               // 0..56

  v8f acc[4];
  float mrow[8], lrow[8], scr[8];
#pragma unroll
  for (int f = 0; f < 4; ++f) acc[f] = (v8f)(0.0f);
#pragma unroll
  for (int r = 0; r < 8; ++r) { mrow[r] = -1e30f; lrow[r] = 0.0f; }

  for (int kt = 0; kt < TT; kt += 32) {
    __syncthreads();
    // stage V tile transposed: vT[dv][key], 16B global loads
    {
      v8us vv = *(const v8us*)(Vb + base + (size_t)(kt + vkey) * HH + vdv8);
#pragma unroll
      for (int j = 0; j < 8; ++j) vT[(vdv8 + j) * 32 + vkey] = vv[j];
    }
    if (kt + 32 < TT) {
      __builtin_prefetch(Vb + base + (size_t)(kt + 32 + vkey) * HH + vdv8, 0, 0);
      __builtin_prefetch(Kb + base + (size_t)(kt + 32 + lh) * HH + lq * 16, 0, 0);
    }
    __syncthreads();

    // S = q . k^T  (k^T B-fragments are contiguous d-loads from global)
    v8f c01[2];
#pragma unroll
    for (int half = 0; half < 2; ++half) {
      const unsigned short* kp =
          Kb + base + (size_t)(kt + half * 16 + lh) * HH + lq * 8;
      v16bf kb0 = mk16(ld8(kp), ld8(kp + 16));        // d 0..31
      v16bf kb1 = mk16(ld8(kp + 32), ld8(kp + 48));   // d 32..63
      v8f c = (v8f)(0.0f);
      c = wmma_bf16(aq0, kb0, c);
      c = wmma_bf16(aq1, kb1, c);
      c01[half] = c;
    }

    // online softmax: row max via 16-lane shuffles; row sum via WMMA below
#pragma unroll
    for (int r = 0; r < 8; ++r) {
      float s0 = c01[0][r] * 0.125f;   // 1/sqrt(64)
      float s1 = c01[1][r] * 0.125f;
      float tm = fmaxf(s0, s1);
#pragma unroll
      for (int m = 1; m < 16; m <<= 1) tm = fmaxf(tm, __shfl_xor(tm, m, 32));
      float mn = fmaxf(mrow[r], tm);
      float sc = __expf(mrow[r] - mn);
      scr[r] = sc;
      mrow[r] = mn;
      float p0 = __expf(s0 - mn);
      float p1 = __expf(s1 - mn);
#pragma unroll
      for (int f = 0; f < 4; ++f) acc[f][r] = acc[f][r] * sc;
      int prow = r + lq * 8;
      pbuf[wid * 512 + prow * 32 + lh] = f2b(p0);
      pbuf[wid * 512 + prow * 32 + 16 + lh] = f2b(p1);
    }
    __syncthreads();

    // O += P . V ; row sums = P . ones (broadcast across lanes in C layout)
    const unsigned short* pp = &pbuf[wid * 512 + lh * 32 + lq * 8];
    v16bf pA = mk16(ld8(pp), ld8(pp + 16));
    v8f rsum = wmma_bf16(pA, onesB, (v8f)(0.0f));
#pragma unroll
    for (int r = 0; r < 8; ++r) lrow[r] = lrow[r] * scr[r] + rsum[r];
#pragma unroll
    for (int f = 0; f < 4; ++f) {
      const unsigned short* vp = &vT[(f * 16 + lh) * 32 + lq * 8];
      v16bf vB = mk16(ld8(vp), ld8(vp + 16));
      acc[f] = wmma_bf16(pA, vB, acc[f]);
    }
  }

  // epilogue: normalize and store bf16
#pragma unroll
  for (int f = 0; f < 4; ++f)
#pragma unroll
    for (int r = 0; r < 8; ++r) {
      int row = q0 + r + lq * 8;
      int dv = f * 16 + lh;
      O[base + (size_t)row * HH + dv] = f2b(acc[f][r] / lrow[r]);
    }
}

// ---------------------------------------------------------------------------
// BatchNorm batch-stats folded into per-channel affine: y = x*sa + sb
// ---------------------------------------------------------------------------
__global__ __launch_bounds__(256)
void bn_stats(const float* __restrict__ Wx, const float* __restrict__ gamma,
              const float* __restrict__ beta, float* __restrict__ sab) {
  __shared__ float ssum[256], ssq[256];
  const int h = blockIdx.x;
  float s = 0.0f, q = 0.0f;
  for (int r = threadIdx.x; r < MT; r += 256) {
    float x = Wx[(size_t)r * HH + h];
    s += x; q += x * x;
  }
  ssum[threadIdx.x] = s; ssq[threadIdx.x] = q;
  __syncthreads();
  for (int off = 128; off > 0; off >>= 1) {
    if (threadIdx.x < off) {
      ssum[threadIdx.x] += ssum[threadIdx.x + off];
      ssq[threadIdx.x] += ssq[threadIdx.x + off];
    }
    __syncthreads();
  }
  if (threadIdx.x == 0) {
    const float inv_n = 1.0f / (float)MT;
    float mean = ssum[0] * inv_n;
    float var = ssq[0] * inv_n - mean * mean;
    float rstd = rsqrtf(var + 1e-5f);
    float sa = gamma[h] * rstd;
    sab[h] = sa;
    sab[HH + h] = beta[h] - mean * sa;
  }
}

// ---------------------------------------------------------------------------
// LIF scan: one thread per (b,h) channel, sequential over T.
// u = 0.5*u + wx ; s = (u > 1) ; u = (1-s)*u
// ---------------------------------------------------------------------------
__global__ __launch_bounds__(256)
void lif_scan_k(const float* __restrict__ Wx, const float* __restrict__ sab,
                float* __restrict__ out) {
  const int ch = blockIdx.x * 256 + threadIdx.x;   // 0..B*H-1
  const int b = ch / HH, hh = ch % HH;
  const float sa = sab[hh], sb = sab[HH + hh];
  float u = 0.0f;
  const size_t base = (size_t)b * TT * HH + hh;
  for (int t = 0; t < TT; ++t) {
    float wx = Wx[base + (size_t)t * HH] * sa + sb;
    u = 0.5f * u + wx;
    float s = (u > 1.0f) ? 1.0f : 0.0f;
    out[base + (size_t)t * HH] = s;
    u = (1.0f - s) * u;
  }
}

// ---------------------------------------------------------------------------
// Launch: qkv projections -> flash attention -> Wo(+res) -> W -> BN -> LIF
// ---------------------------------------------------------------------------
extern "C" void kernel_launch(void* const* d_in, const int* in_sizes, int n_in,
                              void* d_out, int out_size, void* d_ws,
                              size_t ws_size, hipStream_t stream) {
  const float* v  = (const float*)d_in[0];
  const float* a  = (const float*)d_in[1];
  const float* Wq = (const float*)d_in[2];
  const float* Wk = (const float*)d_in[3];
  const float* Wv = (const float*)d_in[4];
  const float* Wo = (const float*)d_in[5];
  const float* W  = (const float*)d_in[6];
  const float* g  = (const float*)d_in[7];
  const float* be = (const float*)d_in[8];

  char* ws = (char*)d_ws;
  const size_t szbf = (size_t)MT * HH * sizeof(unsigned short);  // 8 MB each
  unsigned short* qbf    = (unsigned short*)(ws + 0 * szbf);
  unsigned short* kbf    = (unsigned short*)(ws + 1 * szbf);
  unsigned short* valbf  = (unsigned short*)(ws + 2 * szbf);
  unsigned short* attnbf = (unsigned short*)(ws + 3 * szbf);
  unsigned short* xbf    = (unsigned short*)(ws + 4 * szbf);
  float* Wx  = (float*)(ws + 5 * szbf);                          // 16 MB
  float* sab = (float*)(ws + 5 * szbf + (size_t)MT * HH * sizeof(float));

  dim3 gg(HH / 128, MT / 128);  // (4, 64)

  // q/k/v projections (fp32 in -> bf16 out)
  gemm_bf16_wmma<0,0,0,1><<<gg, 256, 0, stream>>>(v, Wq, nullptr, nullptr, qbf,   HH, HH);
  gemm_bf16_wmma<0,0,0,1><<<gg, 256, 0, stream>>>(a, Wk, nullptr, nullptr, kbf,   HH, HH);
  gemm_bf16_wmma<0,0,0,1><<<gg, 256, 0, stream>>>(a, Wv, nullptr, nullptr, valbf, HH, HH);

  // attention core
  flash_attn_wmma<<<dim3(TT / 128, NHD, BB), 256, 0, stream>>>(qbf, kbf, valbf, attnbf);

  // x = attn @ Wo + a   (bf16 out for next gemm)
  gemm_bf16_wmma<1,1,0,1><<<gg, 256, 0, stream>>>(attnbf, Wo, a, nullptr, xbf, HH, HH);

  // Wx = x @ W   (fp32 out for BN/LIF)
  gemm_bf16_wmma<1,0,1,0><<<gg, 256, 0, stream>>>(xbf, W, nullptr, Wx, nullptr, HH, HH);

  // BN stats -> per-channel affine
  bn_stats<<<HH, 256, 0, stream>>>(Wx, g, be, sab);

  // sequential LIF over T, coalesced across channels
  lif_scan_k<<<(BB * HH) / 256, 256, 0, stream>>>(Wx, sab, (float*)d_out);
}
